// Encoder_51144470561484
// MI455X (gfx1250) — matrix operand order repaired and checked by reading, not compiled, not run on hardware
//
#include <hip/hip_runtime.h>
#include <hip/hip_bf16.h>
#include <stdint.h>

// Problem constants (match the reference).
#define L_NODES   259200
#define H3_NODES  41162
#define N_NODES   (L_NODES + H3_NODES)
#define LN_EPS    1e-5f

typedef __attribute__((ext_vector_type(16))) __bf16       v16bf;
typedef __attribute__((ext_vector_type(8)))  float        v8f;
typedef __attribute__((ext_vector_type(8)))  unsigned int v8u;

// float -> bf16 with round-to-nearest-even, packed pair into a dword.
__device__ __forceinline__ unsigned short f2bf(float f) {
  union { float f; unsigned int u; } c; c.f = f;
  unsigned int u = c.u;
  unsigned int r = u + 0x7FFFu + ((u >> 16) & 1u);
  return (unsigned short)(r >> 16);
}
__device__ __forceinline__ unsigned int pack2(float lo, float hi) {
  return (unsigned int)f2bf(lo) | ((unsigned int)f2bf(hi) << 16);
}

enum { MODE_PLAIN = 0, MODE_EP1 = 1, MODE_NPR1 = 2 };

// LDS strides (dwords). 20 keeps every fragment base 16B-aligned for
// ds_load_b128 (80*n mod 16 == 0) and spreads rows/cols across banks
// (gcd(20,64)=4 -> 16 distinct step-4 bank residues).
#define ASTRIDE 20
#define BSTRIDE 20

// ---------------------------------------------------------------------------
// Tiled GEMM: out[M,256] = act(Astage[M,Kin] @ W[Kin,256] + bias)
// Block = 256 threads (8 waves), tile = 64 rows x 256 cols.
// K stepped by 32; A/B staged to LDS as bf16 pairs in *fragment-major*
// order so each V_WMMA_F32_16X16X32_BF16 operand is two ds_load_b128.
// Single LDS buffer + register prefetch: global loads for step s+1 are
// issued before the WMMAs of step s so VMEM latency hides under matrix ops.
// MODE selects how the A operand is gathered (fused concat for ep/npr).
// ---------------------------------------------------------------------------
template <int MODE, bool RELU>
__global__ __launch_bounds__(256) void gemm_kernel(
    const float* __restrict__ A, const float* __restrict__ W,
    const float* __restrict__ bias, float* __restrict__ out, int M, int Kin,
    const float* __restrict__ X, const int* __restrict__ DST,
    const float* __restrict__ EA, const float* __restrict__ AGG,
    int rowOffset) {
  // A pairs: [row 0..63][pair 0..15], stride ASTRIDE dwords.
  __shared__ __align__(16) unsigned int Alds[64 * ASTRIDE];
  // B pairs: [col 0..255][pair 0..15], stride BSTRIDE dwords.
  __shared__ __align__(16) unsigned int Blds[256 * BSTRIDE];

  const int tid   = threadIdx.x;
  const int lane  = tid & 31;
  const int wave  = tid >> 5;
  const int half  = lane >> 4;
  const int l16   = lane & 15;
  const int waveM = wave & 3;   // which 16-row slice of the 64-row tile
  const int waveN = wave >> 2;  // which 128-col half
  const int blockM = blockIdx.x * 64;

  v8f acc[8] = {};

  const int ksteps = (Kin + 31) / 32;
  // A staging: thread -> (row 0..63, 4 consecutive pairs)
  const int arow = tid >> 2;
  const int apq  = (tid & 3) * 4;
  // B staging: thread -> (kpair 0..15, 16 consecutive cols)
  const int bkp = tid >> 4;
  const int bc0 = (tid & 15) * 16;

  // Register prefetch buffers.
  float aval[8];
  float blo[16], bhi[16];

  // ---- global -> registers for K-step s ----
  auto loadA = [&](int s) {
    const int k0 = s * 32;
    const int grow = blockM + arow;
#pragma unroll
    for (int j = 0; j < 8; ++j) {
      const int gk = k0 + apq * 2 + j;
      float v = 0.0f;
      if (grow < M && gk < Kin) {
        if (MODE == MODE_PLAIN) {
          v = A[(long long)grow * Kin + gk];
        } else if (MODE == MODE_EP1) {
          if (gk < 256)      v = X[(long long)grow * 256 + gk];
          else if (gk < 512) v = X[(long long)DST[grow] * 256 + (gk - 256)];
          else               v = EA[(long long)grow * 2 + (gk - 512)];
        } else {  // MODE_NPR1
          const long long r = (long long)rowOffset + grow;
          if (gk < 256) v = X[r * 256 + gk];
          else          v = AGG[r * 2 + (gk - 256)];
        }
      }
      aval[j] = v;
    }
  };
  auto loadB = [&](int s) {
    const int k0 = s * 32;
    const int ke = k0 + bkp * 2, ko = ke + 1;
    if (ke < Kin) {
      const float4* p = (const float4*)(W + (long long)ke * 256 + bc0);
#pragma unroll
      for (int q = 0; q < 4; ++q) {
        float4 f = p[q];
        blo[4 * q + 0] = f.x; blo[4 * q + 1] = f.y;
        blo[4 * q + 2] = f.z; blo[4 * q + 3] = f.w;
      }
    } else {
#pragma unroll
      for (int i = 0; i < 16; ++i) blo[i] = 0.0f;
    }
    if (ko < Kin) {
      const float4* p = (const float4*)(W + (long long)ko * 256 + bc0);
#pragma unroll
      for (int q = 0; q < 4; ++q) {
        float4 f = p[q];
        bhi[4 * q + 0] = f.x; bhi[4 * q + 1] = f.y;
        bhi[4 * q + 2] = f.z; bhi[4 * q + 3] = f.w;
      }
    } else {
#pragma unroll
      for (int i = 0; i < 16; ++i) bhi[i] = 0.0f;
    }
  };
  // ---- registers -> LDS (bf16 pairs, fragment-major) ----
  auto storeA = [&]() {
#pragma unroll
    for (int j = 0; j < 4; ++j)
      Alds[arow * ASTRIDE + apq + j] = pack2(aval[2 * j], aval[2 * j + 1]);
  };
  auto storeB = [&]() {
#pragma unroll
    for (int i = 0; i < 16; ++i)
      Blds[(bc0 + i) * BSTRIDE + bkp] = pack2(blo[i], bhi[i]);
  };

  loadA(0);
  loadB(0);
  for (int s = 0; s < ksteps; ++s) {
    storeA();
    storeB();
    __syncthreads();
    if (s + 1 < ksteps) {  // prefetch next K-step while WMMAs run
      loadA(s + 1);
      loadB(s + 1);
    }

    // A fragment: pairs {4h..4h+3} and {8+4h..8+4h+3} -> two b128 loads.
    const unsigned int* arow_p = &Alds[(waveM * 16 + l16) * ASTRIDE];
    const uint4 a0 = *(const uint4*)(arow_p + 4 * half);
    const uint4 a1 = *(const uint4*)(arow_p + 8 + 4 * half);
    v8u au;
    au[0] = a0.x; au[1] = a0.y; au[2] = a0.z; au[3] = a0.w;
    au[4] = a1.x; au[5] = a1.y; au[6] = a1.z; au[7] = a1.w;
    const v16bf av = __builtin_bit_cast(v16bf, au);

#pragma unroll
    for (int t = 0; t < 8; ++t) {
      const int n = waveN * 128 + t * 16 + l16;
      // B fragment: pairs {4h..4h+3} and {8+4h..8+4h+3} -> two b128 loads.
      const unsigned int* bcol_p = &Blds[n * BSTRIDE + 4 * half];
      const uint4 b0 = *(const uint4*)(bcol_p);
      const uint4 b1 = *(const uint4*)(bcol_p + 8);
      v8u bu;
      bu[0] = b0.x; bu[1] = b0.y; bu[2] = b0.z; bu[3] = b0.w;
      bu[4] = b1.x; bu[5] = b1.y; bu[6] = b1.z; bu[7] = b1.w;
      const v16bf bv = __builtin_bit_cast(v16bf, bu);
      acc[t] = __builtin_amdgcn_wmma_f32_16x16x32_bf16(
          false, av, false, bv, (short)0, acc[t], false, false);
    }
    __syncthreads();
  }

  // ---- epilogue: D layout (ISA 7.12.2): row = v + 8*half, col = l16 ----
#pragma unroll
  for (int t = 0; t < 8; ++t) {
    const int col = waveN * 128 + t * 16 + l16;
    const float b = bias[col];
#pragma unroll
    for (int v = 0; v < 8; ++v) {
      const int row = blockM + waveM * 16 + v + 8 * half;
      if (row < M) {
        float val = acc[t][v] + b;
        if (RELU) val = fmaxf(val, 0.0f);
        out[(long long)row * 256 + col] = val;
      }
    }
  }
}

// ---------------------------------------------------------------------------
// Row LayerNorm over 256 features, optional residual add, one wave per row.
// ---------------------------------------------------------------------------
__global__ __launch_bounds__(256) void layernorm_kernel(
    const float* __restrict__ in, const float* __restrict__ gamma,
    const float* __restrict__ beta, const float* res, float* out, int M) {
  const int lane = threadIdx.x & 31;
  const int waveId = (blockIdx.x * blockDim.x + threadIdx.x) >> 5;
  const int nWaves = (gridDim.x * blockDim.x) >> 5;
  for (int row = waveId; row < M; row += nWaves) {
    float v[8];
    float s = 0.0f, s2 = 0.0f;
#pragma unroll
    for (int j = 0; j < 8; ++j) {
      v[j] = in[(long long)row * 256 + j * 32 + lane];
      s += v[j];
      s2 += v[j] * v[j];
    }
#pragma unroll
    for (int off = 16; off > 0; off >>= 1) {
      s  += __shfl_xor(s, off, 32);
      s2 += __shfl_xor(s2, off, 32);
    }
    const float mu  = s * (1.0f / 256.0f);
    const float var = s2 * (1.0f / 256.0f) - mu * mu;
    const float rs  = rsqrtf(var + LN_EPS);
#pragma unroll
    for (int j = 0; j < 8; ++j) {
      const int c = j * 32 + lane;
      float o = (v[j] - mu) * rs * gamma[c] + beta[c];
      if (res) o += res[(long long)row * 256 + c];
      out[(long long)row * 256 + c] = o;
    }
  }
}

// Edge-encoder layer 1 (K=2 -> 256, ReLU): trivially memory bound, VALU.
__global__ void ee1_kernel(const float* __restrict__ dists,
                           const float* __restrict__ W,
                           const float* __restrict__ b, float* __restrict__ out,
                           int E) {
  long long i = (long long)blockIdx.x * blockDim.x + threadIdx.x;
  if (i >= (long long)E * 256) return;
  const long long e = i >> 8;
  const int n = (int)(i & 255);
  const float d0 = dists[e * 2], d1 = dists[e * 2 + 1];
  out[i] = fmaxf(d0 * W[n] + d1 * W[256 + n] + b[n], 0.0f);
}

// MLP head 256 -> 2, + LayerNorm(2), + optional residual. One wave per row.
__global__ __launch_bounds__(256) void head_kernel(
    const float* __restrict__ in, const float* __restrict__ W2,
    const float* __restrict__ b2, const float* __restrict__ gamma,
    const float* __restrict__ beta, const float* res, float* out, int M) {
  const int lane = threadIdx.x & 31;
  const int waveId = (blockIdx.x * blockDim.x + threadIdx.x) >> 5;
  const int nWaves = (gridDim.x * blockDim.x) >> 5;
  for (int row = waveId; row < M; row += nWaves) {
    float a0 = 0.0f, a1 = 0.0f;
#pragma unroll
    for (int j = 0; j < 8; ++j) {
      const int k = j * 32 + lane;
      const float h = in[(long long)row * 256 + k];
      a0 += h * W2[k * 2];
      a1 += h * W2[k * 2 + 1];
    }
#pragma unroll
    for (int off = 16; off > 0; off >>= 1) {
      a0 += __shfl_xor(a0, off, 32);
      a1 += __shfl_xor(a1, off, 32);
    }
    if (lane == 0) {
      const float v0 = a0 + b2[0], v1 = a1 + b2[1];
      const float mu = 0.5f * (v0 + v1);
      const float var =
          0.5f * ((v0 - mu) * (v0 - mu) + (v1 - mu) * (v1 - mu));
      const float rs = rsqrtf(var + LN_EPS);
      float o0 = (v0 - mu) * rs * gamma[0] + beta[0];
      float o1 = (v1 - mu) * rs * gamma[1] + beta[1];
      if (res) {
        o0 += res[(long long)row * 2];
        o1 += res[(long long)row * 2 + 1];
      }
      out[(long long)row * 2]     = o0;
      out[(long long)row * 2 + 1] = o1;
    }
  }
}

// Scatter-sum edges into target nodes (global_atomic_add_f32).
__global__ void scatter_kernel(const float* __restrict__ ea,
                               const int* __restrict__ dst,
                               float* __restrict__ agg, int E) {
  const int e = blockIdx.x * blockDim.x + threadIdx.x;
  if (e >= E) return;
  const int d = dst[e];
  atomicAdd(&agg[(long long)d * 2],     ea[(long long)e * 2]);
  atomicAdd(&agg[(long long)d * 2 + 1], ea[(long long)e * 2 + 1]);
}

extern "C" void kernel_launch(void* const* d_in, const int* in_sizes, int n_in,
                              void* d_out, int out_size, void* d_ws,
                              size_t ws_size, hipStream_t stream) {
  (void)in_sizes; (void)n_in; (void)out_size; (void)ws_size;
  const float* features = (const float*)d_in[0];
  const float* dists    = (const float*)d_in[1];
  const int*   targets  = (const int*)d_in[2];
  // ne
  const float *neW0 = (const float*)d_in[3],  *neW1 = (const float*)d_in[4],
              *neW2 = (const float*)d_in[5],  *neB0 = (const float*)d_in[6],
              *neB1 = (const float*)d_in[7],  *neB2 = (const float*)d_in[8],
              *neG  = (const float*)d_in[9],  *neBt = (const float*)d_in[10];
  // ee
  const float *eeW0 = (const float*)d_in[11], *eeW1 = (const float*)d_in[12],
              *eeW2 = (const float*)d_in[13], *eeB0 = (const float*)d_in[14],
              *eeB1 = (const float*)d_in[15], *eeB2 = (const float*)d_in[16],
              *eeG  = (const float*)d_in[17], *eeBt = (const float*)d_in[18];
  // ep
  const float *epW0 = (const float*)d_in[19], *epW1 = (const float*)d_in[20],
              *epW2 = (const float*)d_in[21], *epB0 = (const float*)d_in[22],
              *epB1 = (const float*)d_in[23], *epB2 = (const float*)d_in[24],
              *epG  = (const float*)d_in[25], *epBt = (const float*)d_in[26];
  // npr
  const float *npW0 = (const float*)d_in[27], *npW1 = (const float*)d_in[28],
              *npW2 = (const float*)d_in[29], *npB0 = (const float*)d_in[30],
              *npB1 = (const float*)d_in[31], *npB2 = (const float*)d_in[32],
              *npG  = (const float*)d_in[33], *npBt = (const float*)d_in[34];

  // Workspace layout (f32): x[N,256] | h1[L,256] | h2[L,256] | ea[L,2] | agg[N,2]
  char* ws = (char*)d_ws;
  float* x  = (float*)ws;
  size_t off = (size_t)N_NODES * 256 * 4;
  float* h1 = (float*)(ws + off); off += (size_t)L_NODES * 256 * 4;
  float* h2 = (float*)(ws + off); off += (size_t)L_NODES * 256 * 4;
  float* ea = (float*)(ws + off); off += (size_t)L_NODES * 2 * 4;
  float* agg = (float*)(ws + off);

  const dim3 blk(256);
  const int gL = (L_NODES + 63) / 64;
  const int gH = (H3_NODES + 63) / 64;

  // h3 node states are zero-initialized; zero the aggregation buffer too.
  hipMemsetAsync(x + (size_t)L_NODES * 256, 0, (size_t)H3_NODES * 256 * 4,
                 stream);
  hipMemsetAsync(agg, 0, (size_t)N_NODES * 2 * 4, stream);

  // ---- node encoder: features[L,78] -> x[0:L,256] ----
  gemm_kernel<MODE_PLAIN, true><<<gL, blk, 0, stream>>>(
      features, neW0, neB0, h1, L_NODES, 78, nullptr, nullptr, nullptr,
      nullptr, 0);
  gemm_kernel<MODE_PLAIN, true><<<gL, blk, 0, stream>>>(
      h1, neW1, neB1, h2, L_NODES, 256, nullptr, nullptr, nullptr, nullptr, 0);
  gemm_kernel<MODE_PLAIN, false><<<gL, blk, 0, stream>>>(
      h2, neW2, neB2, h1, L_NODES, 256, nullptr, nullptr, nullptr, nullptr, 0);
  layernorm_kernel<<<2048, blk, 0, stream>>>(h1, neG, neBt, nullptr, x,
                                             L_NODES);

  // ---- edge encoder: dists[L,2] -> ea[L,2] ----
  const long long totEE1 = (long long)L_NODES * 256;
  ee1_kernel<<<(int)((totEE1 + 255) / 256), blk, 0, stream>>>(dists, eeW0,
                                                              eeB0, h1,
                                                              L_NODES);
  gemm_kernel<MODE_PLAIN, true><<<gL, blk, 0, stream>>>(
      h1, eeW1, eeB1, h2, L_NODES, 256, nullptr, nullptr, nullptr, nullptr, 0);
  head_kernel<<<2048, blk, 0, stream>>>(h2, eeW2, eeB2, eeG, eeBt, nullptr, ea,
                                        L_NODES);

  // ---- edge processor: cat[x[src], x[dst], ea] -> ea (+residual) ----
  gemm_kernel<MODE_EP1, true><<<gL, blk, 0, stream>>>(
      nullptr, epW0, epB0, h1, L_NODES, 514, x, targets, ea, nullptr, 0);
  gemm_kernel<MODE_PLAIN, true><<<gL, blk, 0, stream>>>(
      h1, epW1, epB1, h2, L_NODES, 256, nullptr, nullptr, nullptr, nullptr, 0);
  head_kernel<<<2048, blk, 0, stream>>>(h2, epW2, epB2, epG, epBt, ea, ea,
                                        L_NODES);

  // ---- scatter-sum edges to target nodes ----
  scatter_kernel<<<(L_NODES + 255) / 256, blk, 0, stream>>>(ea, targets, agg,
                                                            L_NODES);

  // ---- node processor on h3 rows only; write result straight to d_out ----
  gemm_kernel<MODE_NPR1, true><<<gH, blk, 0, stream>>>(
      nullptr, npW0, npB0, h1, H3_NODES, 258, x, nullptr, nullptr, agg,
      L_NODES);
  gemm_kernel<MODE_PLAIN, true><<<gH, blk, 0, stream>>>(
      h1, npW1, npB1, h2, H3_NODES, 256, nullptr, nullptr, nullptr, nullptr,
      0);
  gemm_kernel<MODE_PLAIN, false><<<gH, blk, 0, stream>>>(
      h2, npW2, npB2, h1, H3_NODES, 256, nullptr, nullptr, nullptr, nullptr,
      0);
  layernorm_kernel<<<1024, blk, 0, stream>>>(
      h1, npG, npBt, x + (size_t)L_NODES * 256, (float*)d_out, H3_NODES);
}
